// RelationBlock_89060441850130
// MI455X (gfx1250) — matrix operand order repaired and checked by reading, not compiled, not run on hardware
//
#include <hip/hip_runtime.h>
#include <hip/hip_bf16.h>
#include <math.h>

#define GATE_DIM 1935
#define NROW 384
#define D1 300
#define OUTD 300
#define KPAD 320
#define NPAD 304
#define MT_PER_BLOCK 4

typedef __bf16 bf16_t;
typedef __attribute__((ext_vector_type(16))) __bf16 v16bf;
typedef __attribute__((ext_vector_type(8)))  __bf16 v8bf;
typedef __attribute__((ext_vector_type(8)))  float  v8f;

// ---------------------------------------------------------------------------
// P1: fAll[v][c] = ((feature_v * sigmoid(gate)) @ E)[c], v in [0,768)
// 4 rows per block staged in LDS; E reads coalesced & reused 4x.
// ---------------------------------------------------------------------------
__global__ __launch_bounds__(320) void k_gated_proj(
    const float* __restrict__ feat1, const float* __restrict__ feat2,
    const float* __restrict__ gate1, const float* __restrict__ gate2,
    const float* __restrict__ E, float* __restrict__ fAll)
{
    __shared__ float xr[4][GATE_DIM];
    const int tid = threadIdx.x;
    const int rb  = blockIdx.x;                    // 0..191
    for (int j = 0; j < 4; ++j) {
        const int v = rb * 4 + j;
        const float* feat = (v < NROW) ? feat1 : feat2;
        const float* gate = (v < NROW) ? gate1 : gate2;
        const int r = (v < NROW) ? v : v - NROW;
        for (int k = tid; k < GATE_DIM; k += 320) {
            float g = 1.0f / (1.0f + __expf(-gate[k]));
            xr[j][k] = feat[r * GATE_DIM + k] * g;
        }
    }
    __syncthreads();
    if (tid < D1) {
        float a0 = 0.f, a1 = 0.f, a2 = 0.f, a3 = 0.f;
        for (int k = 0; k < GATE_DIM; ++k) {
            const float e = E[k * D1 + tid];
            a0 = fmaf(xr[0][k], e, a0);
            a1 = fmaf(xr[1][k], e, a1);
            a2 = fmaf(xr[2][k], e, a2);
            a3 = fmaf(xr[3][k], e, a3);
        }
        const int v0 = rb * 4;
        fAll[(v0 + 0) * D1 + tid] = a0;
        fAll[(v0 + 1) * D1 + tid] = a1;
        fAll[(v0 + 2) * D1 + tid] = a2;
        fAll[(v0 + 3) * D1 + tid] = a3;
    }
}

// ---------------------------------------------------------------------------
// P0: transpose fc1_w [300][600] into WaT[o][c]=fc1_w[c][o], WbT[o][c]=fc1_w[c][300+o]
// ---------------------------------------------------------------------------
__global__ void k_transpose_fc1(const float* __restrict__ fc1_w,
                                float* __restrict__ WaT, float* __restrict__ WbT)
{
    const int idx = blockIdx.x * blockDim.x + threadIdx.x;
    if (idx >= 2 * D1 * D1) return;
    const int d   = idx / (D1 * D1);
    const int rem = idx - d * (D1 * D1);
    const int o = rem / D1;
    const int c = rem - o * D1;
    const float v = fc1_w[c * 600 + d * D1 + o];
    (d ? WbT : WaT)[rem] = v;
}

// ---------------------------------------------------------------------------
// P2: SAll[v][c] = sum_o fAll[v][o] * WT[o][c]   (WT=WaT for v<384 else WbT)
// ---------------------------------------------------------------------------
__global__ __launch_bounds__(320) void k_semantic_proj(
    const float* __restrict__ fAll, const float* __restrict__ WaT,
    const float* __restrict__ WbT, float* __restrict__ SAll)
{
    __shared__ float fr[4][D1];
    const int tid = threadIdx.x;
    const int rb  = blockIdx.x;                    // 0..191 ; 384%4==0 -> uniform side
    for (int j = 0; j < 4; ++j)
        for (int k = tid; k < D1; k += 320) fr[j][k] = fAll[(rb * 4 + j) * D1 + k];
    __syncthreads();
    if (tid < D1) {
        const float* WT = (rb * 4 < NROW) ? WaT : WbT;
        float a0 = 0.f, a1 = 0.f, a2 = 0.f, a3 = 0.f;
        for (int o = 0; o < D1; ++o) {
            const float w = WT[o * D1 + tid];
            a0 = fmaf(fr[0][o], w, a0);
            a1 = fmaf(fr[1][o], w, a1);
            a2 = fmaf(fr[2][o], w, a2);
            a3 = fmaf(fr[3][o], w, a3);
        }
        const int v0 = rb * 4;
        SAll[(v0 + 0) * D1 + tid] = a0;
        SAll[(v0 + 1) * D1 + tid] = a1;
        SAll[(v0 + 2) * D1 + tid] = a2;
        SAll[(v0 + 3) * D1 + tid] = a3;
    }
}

// ---------------------------------------------------------------------------
// P3: spatial projections (K=5): T1[r][c]=box1[r]@fc2_w[c][0:5], T2 uses cols 5:10
// ---------------------------------------------------------------------------
__global__ void k_spatial_proj(const float* __restrict__ box1, const float* __restrict__ box2,
                               const float* __restrict__ fc2_w,
                               float* __restrict__ T1, float* __restrict__ T2)
{
    const int idx = blockIdx.x * blockDim.x + threadIdx.x;
    if (idx >= 2 * NROW * D1) return;
    const int d   = idx / (NROW * D1);
    const int rem = idx - d * (NROW * D1);
    const int r = rem / D1;
    const int c = rem - r * D1;
    const float* box = d ? box2 : box1;
    float acc = 0.f;
    #pragma unroll
    for (int k = 0; k < 5; ++k) acc = fmaf(box[r * 5 + k], fc2_w[c * 10 + d * 5 + k], acc);
    (d ? T2 : T1)[rem] = acc;
}

// ---------------------------------------------------------------------------
// P4: convert fc3_w [300][300] f32 -> zero-padded bf16 Bbuf [304][320]
// ---------------------------------------------------------------------------
__global__ void k_bconv(const float* __restrict__ fc3_w, bf16_t* __restrict__ Bbuf)
{
    const int idx = blockIdx.x * blockDim.x + threadIdx.x;
    if (idx >= NPAD * KPAD) return;
    const int p = idx / KPAD;
    const int k = idx - p * KPAD;
    const float v = (p < OUTD && k < D1) ? fc3_w[p * D1 + k] : 0.0f;
    Bbuf[idx] = (bf16_t)v;
}

// ---------------------------------------------------------------------------
// Main fused kernel: build mixed A-tile (16 rows x 320 K, bf16) in LDS
// (double-buffered), then D = A x fc3_w^T via v_wmma_f32_16x16x32_bf16.
// 19 waves = 19 N-tiles of 16. B fragments register-resident across M-tiles.
// All 10 A fragments are hoisted into distinct registers before the WMMA
// chain so the ds_load latency is paid once per tile, not per K-step.
// ---------------------------------------------------------------------------
__global__ __launch_bounds__(608) void k_relation_main(
    const float* __restrict__ S1, const float* __restrict__ S2,
    const float* __restrict__ T1, const float* __restrict__ T2,
    const float* __restrict__ b1, const float* __restrict__ b2,
    const float* __restrict__ wloc, const bf16_t* __restrict__ Bbuf,
    const float* __restrict__ b3, float* __restrict__ out)
{
    __shared__ bf16_t Alds[2][16 * KPAD];          // 2 x 10 KB
    const int tid  = threadIdx.x;
    const int wave = tid >> 5;                     // 0..18 -> N-tile
    const int lane = tid & 31;
    const int hi   = lane >> 4;
    const int l15  = lane & 15;

    // softmax over 2 logits
    const float e0 = __expf(wloc[0]);
    const float e1 = __expf(wloc[1]);
    const float w0 = e0 / (e0 + e1);
    const float w1 = e1 / (e0 + e1);

    // --- load register-resident B fragments (column block of 16 p's, K=320) ---
    // B(32x16) layout: lanes 0-15: N=lane, K=0..15 contiguous; lanes 16-31: same N, K=16..31
    v16bf Bfrag[10];
    {
        const bf16_t* bp = Bbuf + (wave * 16 + l15) * KPAD + hi * 16;
        #pragma unroll
        for (int kt = 0; kt < 10; ++kt) {
            v8bf lo  = *(const v8bf*)(bp + kt * 32);
            v8bf hi8 = *(const v8bf*)(bp + kt * 32 + 8);
            Bfrag[kt] = __builtin_shufflevector(lo, hi8,
                0, 1, 2, 3, 4, 5, 6, 7, 8, 9, 10, 11, 12, 13, 14, 15);
        }
    }

    const int   pcol = wave * 16 + l15;
    const float bias = (pcol < OUTD) ? b3[pcol] : 0.0f;
    const int   mtile0 = blockIdx.x * MT_PER_BLOCK;

    // cooperative A-tile build: mixed[nm = mtile*16+r][o] -> bf16 into buffer b
    auto build = [&](int mtile, int b) {
        for (int e = tid; e < 16 * KPAD; e += 608) {
            const int r = e / KPAD;
            const int o = e - r * KPAD;
            float v = 0.0f;
            if (o < D1) {
                const int nm = mtile * 16 + r;
                const int n  = nm / NROW;
                const int m  = nm - n * NROW;
                const float s = S1[n * D1 + o] + S2[m * D1 + o] + b1[o];
                const float t = T1[n * D1 + o] + T2[m * D1 + o] + b2[o];
                v = w0 * fmaxf(s, 0.0f) + w1 * fmaxf(t, 0.0f);
            }
            Alds[b][e] = (bf16_t)v;
        }
    };

    build(mtile0, 0);
    __syncthreads();

    for (int mtl = 0; mtl < MT_PER_BLOCK; ++mtl) {
        const int mtile = mtile0 + mtl;
        const int cur   = mtl & 1;

        // ---- hoist ALL A fragments for this tile into registers ----
        // A(16x32) layout: lane row = l15; halves 0..15 = K[k0+hi*16 ..] contiguous
        const bf16_t* arow = &Alds[cur][0] + l15 * KPAD + hi * 16;
        v16bf Afrag[10];
        #pragma unroll
        for (int kt = 0; kt < 10; ++kt) {
            v8bf a0 = *(const v8bf*)(arow + kt * 32);
            v8bf a1 = *(const v8bf*)(arow + kt * 32 + 8);
            Afrag[kt] = __builtin_shufflevector(a0, a1,
                0, 1, 2, 3, 4, 5, 6, 7, 8, 9, 10, 11, 12, 13, 14, 15);
        }

        // ---- WMMA chain over 10 K-tiles of 32 ----
        v8f acc = {};
        #pragma unroll
        for (int kt = 0; kt < 10; ++kt) {
            acc = __builtin_amdgcn_wmma_f32_16x16x32_bf16(
                false, Afrag[kt], false, Bfrag[kt], (short)0, acc, false, false);
        }

        // ---- overlap: build next M-tile's A into the other LDS buffer
        //      (independent of acc -> its global loads hide the WMMA tail) ----
        if (mtl + 1 < MT_PER_BLOCK) build(mtile + 1, cur ^ 1);

        // ---- store D: VGPR i -> row (i + hi*8), col = pcol ----
        if (pcol < OUTD) {
            #pragma unroll
            for (int i = 0; i < 8; ++i) {
                const int nm = mtile * 16 + i + hi * 8;
                out[(long)nm * OUTD + pcol] = acc[i] + bias;
            }
        }
        __syncthreads();   // buffer cur^1 fully built before next iteration reads it
    }
}

// ---------------------------------------------------------------------------
extern "C" void kernel_launch(void* const* d_in, const int* in_sizes, int n_in,
                              void* d_out, int out_size, void* d_ws, size_t ws_size,
                              hipStream_t stream) {
    const float* feature1 = (const float*)d_in[0];
    const float* box1     = (const float*)d_in[1];
    const float* feature2 = (const float*)d_in[2];
    const float* box2     = (const float*)d_in[3];
    const float* E        = (const float*)d_in[4];
    const float* gate1    = (const float*)d_in[5];
    const float* gate2    = (const float*)d_in[6];
    const float* wloc     = (const float*)d_in[7];
    const float* fc1_w    = (const float*)d_in[8];
    const float* fc1_b    = (const float*)d_in[9];
    const float* fc2_w    = (const float*)d_in[10];
    const float* fc2_b    = (const float*)d_in[11];
    const float* fc3_w    = (const float*)d_in[12];
    const float* fc3_b    = (const float*)d_in[13];
    float* out = (float*)d_out;

    float* ws = (float*)d_ws;
    float*  fAll = ws;                         // 768*300   = 230400 floats
    float*  WaT  = ws + 230400;                // 90000
    float*  WbT  = ws + 320400;                // 90000
    float*  SAll = ws + 410400;                // 768*300
    float*  T1   = ws + 640800;                // 115200
    float*  T2   = ws + 756000;                // 115200
    bf16_t* Bbuf = (bf16_t*)(ws + 871200);     // 304*320 bf16 (32B aligned)

    k_gated_proj<<<192, 320, 0, stream>>>(feature1, feature2, gate1, gate2, E, fAll);
    k_transpose_fc1<<<(2 * D1 * D1 + 255) / 256, 256, 0, stream>>>(fc1_w, WaT, WbT);
    k_semantic_proj<<<192, 320, 0, stream>>>(fAll, WaT, WbT, SAll);
    k_spatial_proj<<<(2 * NROW * D1 + 255) / 256, 256, 0, stream>>>(box1, box2, fc2_w, T1, T2);
    k_bconv<<<(NPAD * KPAD + 255) / 256, 256, 0, stream>>>(fc3_w, Bbuf);

    // 9216 M-tiles / 4 per block = 2304 blocks of 19 waves
    k_relation_main<<<2304, 608, 0, stream>>>(SAll, SAll + 115200, T1, T2,
                                              fc1_b, fc2_b, wloc, Bbuf, fc3_b, out);
}
